// MiniGPT_65953517797907
// MI455X (gfx1250) — matrix operand order repaired and checked by reading, not compiled, not run on hardware
//
#include <hip/hip_runtime.h>
#include <stdint.h>

// ---------------- types ----------------
typedef __bf16 bf16_t;
typedef __attribute__((ext_vector_type(16))) __bf16          v16bf;
typedef __attribute__((ext_vector_type(8)))  float           v8f;
typedef __attribute__((ext_vector_type(8)))  unsigned short  u16x8;
typedef __attribute__((ext_vector_type(4)))  float           f32x4;
typedef __attribute__((ext_vector_type(4)))  int             v4i;

union FragBF { v16bf v; u16x8 h[2]; };

// model constants (fixed by the reference)
constexpr int kV   = 32000;
constexpr int kD   = 512;
constexpr int kH   = 8;
constexpr int kHD  = 64;
constexpr int kL   = 4;
constexpr int kDFF = 2048;
constexpr int kB   = 2;
constexpr int kS   = 2048;
constexpr int kM   = kB * kS;   // 4096 tokens

// gfx1250 async global->LDS DMA (ASYNCcnt path), guarded so we always compile
#if defined(__has_builtin)
#  if __has_builtin(__builtin_amdgcn_global_load_async_to_lds_b128)
#    define USE_ASYNC_STAGE 1
#  endif
#endif
#ifndef USE_ASYNC_STAGE
#  define USE_ASYNC_STAGE 0
#endif

#define AS_GLOBAL __attribute__((address_space(1)))
#define AS_LOCAL  __attribute__((address_space(3)))

__device__ __forceinline__ unsigned short f2bf(float f) {
  union { float f; unsigned u; } c; c.f = f;
  unsigned u = c.u;
  unsigned r = u + 0x7fffu + ((u >> 16) & 1u);   // round-to-nearest-even
  return (unsigned short)(r >> 16);
}

__device__ __forceinline__ v8f wmma_bf16(v16bf a, v16bf b, v8f c) {
  return __builtin_amdgcn_wmma_f32_16x16x32_bf16(false, a, false, b,
                                                 (short)0, c, false, false);
}

// intra-wave LDS store->load ordering (waves diverge, so no block barrier)
__device__ __forceinline__ void wave_lds_fence() {
  asm volatile("s_wait_dscnt 0" ::: "memory");
}
__device__ __forceinline__ void async_fence() {
  asm volatile("s_wait_asynccnt 0" ::: "memory");
}

// ---------------- embedding ----------------
__global__ __launch_bounds__(256)
void embed_kernel(const float* __restrict__ tok_emb,
                  const float* __restrict__ pos_emb,
                  const int*   __restrict__ x,
                  float* __restrict__ h) {
  int i = blockIdx.x * 256 + threadIdx.x;     // over kM*kD
  int t = i >> 9;                             // token index (D=512)
  int d = i & 511;
  int s = t & (kS - 1);
  int tok = x[t];
  h[i] = tok_emb[(size_t)tok * kD + d] + pos_emb[(size_t)s * kD + d];
}

// ---------------- layernorm -> bf16 ----------------
__global__ __launch_bounds__(128)
void ln_kernel(const float* __restrict__ h,
               const float* __restrict__ g,
               const float* __restrict__ b,
               unsigned short* __restrict__ out) {
  __shared__ float red[128];
  int r = blockIdx.x, t = threadIdx.x;
  const float* row = h + (size_t)r * kD;
  float v0 = row[t], v1 = row[t + 128], v2 = row[t + 256], v3 = row[t + 384];
  red[t] = v0 + v1 + v2 + v3;
  __syncthreads();
  #pragma unroll
  for (int o = 64; o > 0; o >>= 1) { if (t < o) red[t] += red[t + o]; __syncthreads(); }
  float mu = red[0] * (1.0f / kD);
  __syncthreads();
  float d0 = v0 - mu, d1 = v1 - mu, d2 = v2 - mu, d3 = v3 - mu;
  red[t] = d0 * d0 + d1 * d1 + d2 * d2 + d3 * d3;
  __syncthreads();
  #pragma unroll
  for (int o = 64; o > 0; o >>= 1) { if (t < o) red[t] += red[t + o]; __syncthreads(); }
  float rs = rsqrtf(red[0] * (1.0f / kD) + 1e-5f);
  unsigned short* orow = out + (size_t)r * kD;
  orow[t]       = f2bf(d0 * rs * g[t]       + b[t]);
  orow[t + 128] = f2bf(d1 * rs * g[t + 128] + b[t + 128]);
  orow[t + 256] = f2bf(d2 * rs * g[t + 256] + b[t + 256]);
  orow[t + 384] = f2bf(d3 * rs * g[t + 384] + b[t + 384]);
}

// ---------------- GEMM: C[M,N] = A[M,K](bf16) * Bw[N,K](f32->bf16)^T ----------------
// epilogue: +bias, optional exact GELU, optional residual add, f32 and/or bf16
// output, optional V-transpose scatter (for the QKV gemm).
template <bool GELU>
__global__ __launch_bounds__(256)
void gemm_bf16_nt(const unsigned short* __restrict__ A,
                  const float* __restrict__ Bw,
                  const float* __restrict__ bias,
                  const float* res,
                  float* outF,
                  unsigned short* __restrict__ outBf,
                  unsigned short* __restrict__ vt,
                  int M, int N, int K) {
  constexpr int BM = 128, BN = 128, BK = 32, STR = 40;   // 40 ushorts/row: 16B-aligned, bank-spread
  __shared__ unsigned short lA[BM * STR];
  __shared__ unsigned short lB[BN * STR];

  const int tid = threadIdx.x, lane = tid & 31, w = tid >> 5;
  const int wm = w & 3, wn = w >> 2;                 // 4x2 wave grid -> 32x64 per wave
  const int bm0 = blockIdx.y * BM, bn0 = blockIdx.x * BN;
  const int lrow = lane & 15, khalf = lane >> 4;
  const int crow = tid >> 1, chalf = tid & 1;        // staging assignment

  v8f acc[2][4];
  #pragma unroll
  for (int m = 0; m < 2; ++m)
    #pragma unroll
    for (int n = 0; n < 4; ++n)
      #pragma unroll
      for (int i = 0; i < 8; ++i) acc[m][n][i] = 0.0f;

  const unsigned short* gA = A + (size_t)(bm0 + crow) * K + chalf * 16;
  const float*          gB = Bw + (size_t)(bn0 + crow) * K + chalf * 16;

  for (int k0 = 0; k0 < K; k0 += BK) {
    __syncthreads();
    // stage A (already bf16): 16 elements / thread
#if USE_ASYNC_STAGE
    __builtin_amdgcn_global_load_async_to_lds_b128(
        (AS_GLOBAL v4i*)(gA + k0),
        (AS_LOCAL v4i*)&lA[crow * STR + chalf * 16], 0, 0);
    __builtin_amdgcn_global_load_async_to_lds_b128(
        (AS_GLOBAL v4i*)(gA + k0 + 8),
        (AS_LOCAL v4i*)&lA[crow * STR + chalf * 16 + 8], 0, 0);
#else
    {
      const u16x8* pa = (const u16x8*)(gA + k0);
      *(u16x8*)&lA[crow * STR + chalf * 16]     = pa[0];
      *(u16x8*)&lA[crow * STR + chalf * 16 + 8] = pa[1];
    }
#endif
    // stage B: load fp32 weights, convert to bf16
    const f32x4* pb = (const f32x4*)(gB + k0);
    float fb[16];
    *(f32x4*)&fb[0]  = pb[0]; *(f32x4*)&fb[4]  = pb[1];
    *(f32x4*)&fb[8]  = pb[2]; *(f32x4*)&fb[12] = pb[3];
    unsigned short ub[16];
    #pragma unroll
    for (int j = 0; j < 16; ++j) ub[j] = f2bf(fb[j]);
    *(u16x8*)&lB[crow * STR + chalf * 16]     = *(u16x8*)&ub[0];
    *(u16x8*)&lB[crow * STR + chalf * 16 + 8] = *(u16x8*)&ub[8];
    // prefetch next K-slice of weights (global_prefetch_b8); L2-resident stream
    if (k0 + BK < K) __builtin_prefetch(gB + k0 + BK, 0, 1);
#if USE_ASYNC_STAGE
    async_fence();
#endif
    __syncthreads();

    FragBF afr[2], bfr[4];
    #pragma unroll
    for (int m = 0; m < 2; ++m) {
      int row = wm * 32 + m * 16 + lrow;
      afr[m].h[0] = *(const u16x8*)&lA[row * STR + khalf * 8];
      afr[m].h[1] = *(const u16x8*)&lA[row * STR + 16 + khalf * 8];
    }
    #pragma unroll
    for (int n = 0; n < 4; ++n) {
      int col = wn * 64 + n * 16 + lrow;
      bfr[n].h[0] = *(const u16x8*)&lB[col * STR + khalf * 8];
      bfr[n].h[1] = *(const u16x8*)&lB[col * STR + 16 + khalf * 8];
    }
    #pragma unroll
    for (int m = 0; m < 2; ++m)
      #pragma unroll
      for (int n = 0; n < 4; ++n)
        acc[m][n] = wmma_bf16(afr[m].v, bfr[n].v, acc[m][n]);
  }

  // epilogue (C layout: VGPR i -> row i+8*(lane>>4), col lane&15)
  #pragma unroll
  for (int m = 0; m < 2; ++m)
    #pragma unroll
    for (int n = 0; n < 4; ++n) {
      int c = bn0 + wn * 64 + n * 16 + lrow;
      float bv = bias ? bias[c] : 0.0f;
      #pragma unroll
      for (int i = 0; i < 8; ++i) {
        int r = bm0 + wm * 32 + m * 16 + i + 8 * khalf;
        float v = acc[m][n][i] + bv;
        if (GELU) v = 0.5f * v * (1.0f + erff(v * 0.70710678118654752f));
        if (res)  v += res[(size_t)r * N + c];
        if (outF)  outF[(size_t)r * N + c] = v;
        if (outBf) outBf[(size_t)r * N + c] = f2bf(v);
        if (vt && c >= 2 * kD) {              // V third of QKV -> vt[b,h,hd,s]
          int bb = r >> 11, ss = r & (kS - 1);
          int e = c - 2 * kD, hh = e >> 6, hd = e & (kHD - 1);
          vt[((((size_t)bb * kH + hh) * kHD + hd) << 11) + ss] = f2bf(v);
        }
      }
    }
}

// ---------------- flash attention (wave = 16 query rows) ----------------
__global__ __launch_bounds__(256)
void attn_kernel(const unsigned short* __restrict__ qkv,  // [kM, 3*kD] bf16
                 const unsigned short* __restrict__ vt,   // [kB,kH,kHD,kS] bf16
                 unsigned short* __restrict__ o) {        // [kM, kD] bf16
  constexpr int STR = 40;
  __shared__ unsigned short P[8][16 * STR];               // per-wave P tile

  const int tid = threadIdx.x, lane = tid & 31, w = tid >> 5;
  const int lrow = lane & 15, khalf = lane >> 4;
  const int bh = blockIdx.y, b = bh >> 3, hh = bh & 7;
  const int q0 = blockIdx.x * 128 + w * 16;               // query tile base (within S)
  const int qtok = b * kS + q0 + lrow;

  // preload Q A-fragments (K = HD = 64 -> two frags)
  FragBF qa[2];
  const unsigned short* qbase = qkv + (size_t)qtok * (3 * kD) + hh * kHD;
  #pragma unroll
  for (int c = 0; c < 2; ++c) {
    qa[c].h[0] = *(const u16x8*)(qbase + c * 32 + khalf * 8);
    qa[c].h[1] = *(const u16x8*)(qbase + c * 32 + khalf * 8 + 16);
  }

  float mrow[8], lrowsum[8];
  v8f Oa[4];
  #pragma unroll
  for (int i = 0; i < 8; ++i) { mrow[i] = -3.0e38f; lrowsum[i] = 0.0f; }
  #pragma unroll
  for (int n = 0; n < 4; ++n)
    #pragma unroll
    for (int i = 0; i < 8; ++i) Oa[n][i] = 0.0f;

  const unsigned short* kbase = qkv + (size_t)(b * kS) * (3 * kD) + kD + hh * kHD;
  const unsigned short* vbase = vt + ((size_t)bh * kHD) * kS;
  const int nkt = (q0 + 16 + 31) >> 5;                    // causal: keys 0..q0+15

  for (int kt = 0; kt < nkt; ++kt) {
    const int kk = kt * 32;
    v8f s0, s1;
    #pragma unroll
    for (int i = 0; i < 8; ++i) { s0[i] = 0.0f; s1[i] = 0.0f; }

    // scores = Q (16xHD) * K^T : two key groups of 16, chained over hd halves
    #pragma unroll
    for (int g = 0; g < 2; ++g) {
      const unsigned short* kp =
          kbase + (size_t)(kk + g * 16 + lrow) * (3 * kD) + khalf * 8;
      FragBF kb0, kb1;
      kb0.h[0] = *(const u16x8*)(kp);       kb0.h[1] = *(const u16x8*)(kp + 16);
      kb1.h[0] = *(const u16x8*)(kp + 32);  kb1.h[1] = *(const u16x8*)(kp + 48);
      v8f sg = g ? s1 : s0;
      sg = wmma_bf16(qa[0].v, kb0.v, sg);
      sg = wmma_bf16(qa[1].v, kb1.v, sg);
      if (g) s1 = sg; else s0 = sg;
    }

    // scale + causal mask + online softmax (rows live across 16-lane halves)
    const bool fullTile = (kk + 31) <= q0;   // wave-uniform: no masking needed
    #pragma unroll
    for (int i = 0; i < 8; ++i) {
      if (fullTile) {
        s0[i] *= 0.125f;
        s1[i] *= 0.125f;
      } else {
        int qr = q0 + i + 8 * khalf;
        s0[i] = ((kk + lrow)      <= qr) ? s0[i] * 0.125f : -3.0e38f;
        s1[i] = ((kk + 16 + lrow) <= qr) ? s1[i] * 0.125f : -3.0e38f;
      }

      float rm = fmaxf(s0[i], s1[i]);
      rm = fmaxf(rm, __shfl_xor(rm, 1, 32));
      rm = fmaxf(rm, __shfl_xor(rm, 2, 32));
      rm = fmaxf(rm, __shfl_xor(rm, 4, 32));
      rm = fmaxf(rm, __shfl_xor(rm, 8, 32));
      float mn = fmaxf(mrow[i], rm);
      float alpha = __expf(mrow[i] - mn);
      float p0 = __expf(s0[i] - mn), p1 = __expf(s1[i] - mn);
      float rs = p0 + p1;
      rs += __shfl_xor(rs, 1, 32);
      rs += __shfl_xor(rs, 2, 32);
      rs += __shfl_xor(rs, 4, 32);
      rs += __shfl_xor(rs, 8, 32);
      lrowsum[i] = lrowsum[i] * alpha + rs;
      mrow[i] = mn;
      #pragma unroll
      for (int n = 0; n < 4; ++n) Oa[n][i] *= alpha;
      s0[i] = p0; s1[i] = p1;
    }

    // bounce P through per-wave LDS: C layout -> A layout
    #pragma unroll
    for (int i = 0; i < 8; ++i) {
      int row = i + 8 * khalf;
      P[w][row * STR + lrow]      = f2bf(s0[i]);
      P[w][row * STR + 16 + lrow] = f2bf(s1[i]);
    }
    wave_lds_fence();
    FragBF pa;
    pa.h[0] = *(const u16x8*)&P[w][lrow * STR + khalf * 8];
    pa.h[1] = *(const u16x8*)&P[w][lrow * STR + 16 + khalf * 8];
    wave_lds_fence();

    // O += P (16x32) * V (32xHD) ; V^T buffer gives contiguous key runs
    #pragma unroll
    for (int n = 0; n < 4; ++n) {
      const unsigned short* vp = vbase + (size_t)(n * 16 + lrow) * kS + kk + khalf * 8;
      FragBF vb;
      vb.h[0] = *(const u16x8*)vp;
      vb.h[1] = *(const u16x8*)(vp + 16);
      Oa[n] = wmma_bf16(pa.v, vb.v, Oa[n]);
    }
  }

  // normalize + store [token, h*HD + hd] bf16
  #pragma unroll
  for (int n = 0; n < 4; ++n)
    #pragma unroll
    for (int i = 0; i < 8; ++i) {
      int token = b * kS + q0 + i + 8 * khalf;
      o[(size_t)token * kD + hh * kHD + n * 16 + lrow] = f2bf(Oa[n][i] / lrowsum[i]);
    }
}

// ---------------- driver ----------------
extern "C" void kernel_launch(void* const* d_in, const int* in_sizes, int n_in,
                              void* d_out, int out_size, void* d_ws, size_t ws_size,
                              hipStream_t stream) {
  const float* tok_emb = (const float*)d_in[0];
  const float* pos_emb = (const float*)d_in[1];
  const float* ln1_g   = (const float*)d_in[2];
  const float* ln1_b   = (const float*)d_in[3];
  const float* qkv_w   = (const float*)d_in[4];
  const float* qkv_b   = (const float*)d_in[5];
  const float* proj_w  = (const float*)d_in[6];
  const float* proj_b  = (const float*)d_in[7];
  const float* ln2_g   = (const float*)d_in[8];
  const float* ln2_b   = (const float*)d_in[9];
  const float* ffn1_w  = (const float*)d_in[10];
  const float* ffn1_b  = (const float*)d_in[11];
  const float* ffn2_w  = (const float*)d_in[12];
  const float* ffn2_b  = (const float*)d_in[13];
  const float* lnf_g   = (const float*)d_in[14];
  const float* lnf_b   = (const float*)d_in[15];
  const float* lm_w    = (const float*)d_in[16];
  const float* lm_b    = (const float*)d_in[17];
  const int*   x       = (const int*)d_in[18];

  // workspace carve-out (~48 MB)
  char* p = (char*)d_ws;
  float* h = (float*)p;               p += (size_t)kM * kD * sizeof(float);
  unsigned short* hn    = (unsigned short*)p;  p += (size_t)kM * kD * 2;
  unsigned short* qkvb  = (unsigned short*)p;  p += (size_t)kM * 3 * kD * 2;
  unsigned short* vtb   = (unsigned short*)p;  p += (size_t)kB * kH * kHD * kS * 2;
  unsigned short* attnb = (unsigned short*)p;  p += (size_t)kM * kD * 2;
  unsigned short* fb    = (unsigned short*)p;  p += (size_t)kM * kDFF * 2;
  (void)ws_size; (void)n_in; (void)in_sizes; (void)out_size;

  embed_kernel<<<kM * kD / 256, 256, 0, stream>>>(tok_emb, pos_emb, x, h);

  for (int l = 0; l < kL; ++l) {
    ln_kernel<<<kM, 128, 0, stream>>>(h, ln1_g + l * kD, ln1_b + l * kD, hn);

    gemm_bf16_nt<false><<<dim3(3 * kD / 128, kM / 128), 256, 0, stream>>>(
        hn, qkv_w + (size_t)l * 3 * kD * kD, qkv_b + (size_t)l * 3 * kD,
        nullptr, nullptr, qkvb, vtb, kM, 3 * kD, kD);

    attn_kernel<<<dim3(kS / 128, kB * kH), 256, 0, stream>>>(qkvb, vtb, attnb);

    gemm_bf16_nt<false><<<dim3(kD / 128, kM / 128), 256, 0, stream>>>(
        attnb, proj_w + (size_t)l * kD * kD, proj_b + (size_t)l * kD,
        h, h, nullptr, nullptr, kM, kD, kD);

    ln_kernel<<<kM, 128, 0, stream>>>(h, ln2_g + l * kD, ln2_b + l * kD, hn);

    gemm_bf16_nt<true><<<dim3(kDFF / 128, kM / 128), 256, 0, stream>>>(
        hn, ffn1_w + (size_t)l * kDFF * kD, ffn1_b + (size_t)l * kDFF,
        nullptr, nullptr, fb, nullptr, kM, kDFF, kD);

    gemm_bf16_nt<false><<<dim3(kD / 128, kM / 128), 256, 0, stream>>>(
        fb, ffn2_w + (size_t)l * kD * kDFF, ffn2_b + (size_t)l * kD,
        h, h, nullptr, nullptr, kM, kD, kDFF);
  }

  ln_kernel<<<kM, 128, 0, stream>>>(h, lnf_g, lnf_b, hn);

  gemm_bf16_nt<false><<<dim3(kV / 128, kM / 128), 256, 0, stream>>>(
      hn, lm_w, lm_b, nullptr, (float*)d_out, nullptr, nullptr, kM, kV, kD);
}
